// InferenceEngine_69715909149464
// MI455X (gfx1250) — matrix operand order repaired and checked by reading, not compiled, not run on hardware
//
#include <hip/hip_runtime.h>
#include <hip/hip_bf16.h>

#define LFSR_MASK 0xFFFFu
#define LFSR_POLY 69643u
#define BLOCK     256
#define NLAYERS   4
#define VOCAB     16384
#define DMODEL    1024
#define BATCH     8
#define TSTEPS    16
#define NSEEDS    (NLAYERS * VOCAB * DMODEL / BLOCK)   // 262144

typedef __attribute__((ext_vector_type(16))) __bf16 v16bf;
typedef __attribute__((ext_vector_type(8)))  float  v8f;

union V16BFCast { uint4 q[2]; v16bf v; };

__device__ __forceinline__ unsigned short f32_to_bf16(float f) {
  unsigned int u = __float_as_uint(f);
  u += 0x7FFFu + ((u >> 16) & 1u);          // round-to-nearest-even
  return (unsigned short)(u >> 16);
}

// ---------------------------------------------------------------------------
// Stage 1: LFSR decompression of all 4 layers into bf16 weights in d_ws.
// One thread per seed; 256 dependent LFSR steps, buffered 8 values (16B) per
// store so the write path uses b128 stores. 134 MB written once; stays in the
// 192 MB L2 so all 16 decode steps stream weights at L2 bandwidth.
// ---------------------------------------------------------------------------
__global__ void __launch_bounds__(256)
decompress_lfsr(const int* __restrict__ seeds, const float* __restrict__ scales,
                unsigned short* __restrict__ W) {
  const int s = blockIdx.x * 256 + threadIdx.x;          // 0..262143
  const float scale = scales[s >> 16];                   // 65536 seeds / layer
  unsigned int state = (unsigned int)seeds[s] & LFSR_MASK;
  unsigned short* base = W + (size_t)s * BLOCK;
  for (int i = 0; i < BLOCK; i += 8) {
    union { unsigned short u16[8]; uint4 q; } tmp;
#pragma unroll
    for (int j = 0; j < 8; ++j) {
      float val = ((float)state * (1.0f / 32768.0f) - 1.0f) * scale;
      tmp.u16[j] = f32_to_bf16(val);
      unsigned int fb = (unsigned int)__popc(state & LFSR_POLY) & 1u;
      state = ((state >> 1) | (fb << 15)) & LFSR_MASK;
    }
    *(uint4*)(base + i) = tmp.q;
  }
}

// ---------------------------------------------------------------------------
// Stage 2a: LayerNorm + fake int8 quantize -> bf16 activations xq[16][1024]
// (rows 8..15 zero padding for the 16-row WMMA tile). One block per row.
// ---------------------------------------------------------------------------
__global__ void __launch_bounds__(256)
ln_quant(const float* __restrict__ xin, unsigned short* __restrict__ xq) {
  const int row = blockIdx.x;            // 0..15
  const int tid = threadIdx.x;
  if (row >= BATCH) {
    for (int i = tid; i < DMODEL; i += 256) xq[row * DMODEL + i] = 0;
    return;
  }
  __shared__ float s1[256], s2[256];
  float x[4];
  float sum = 0.f, sq = 0.f;
#pragma unroll
  for (int k = 0; k < 4; ++k) {
    x[k] = xin[row * DMODEL + tid + k * 256];
    sum += x[k]; sq += x[k] * x[k];
  }
  s1[tid] = sum; s2[tid] = sq;
  __syncthreads();
  for (int off = 128; off > 0; off >>= 1) {
    if (tid < off) { s1[tid] += s1[tid + off]; s2[tid] += s2[tid + off]; }
    __syncthreads();
  }
  const float mu  = s1[0] * (1.f / DMODEL);
  const float var = s2[0] * (1.f / DMODEL) - mu * mu;
  const float rs  = rsqrtf(var + 1e-5f);
  __syncthreads();                       // done reading s1[0]/s2[0]
  float xn[4];
  float amax = 0.f;
#pragma unroll
  for (int k = 0; k < 4; ++k) {
    xn[k] = (x[k] - mu) * rs;
    amax = fmaxf(amax, fabsf(xn[k]));
  }
  s1[tid] = amax;
  __syncthreads();
  for (int off = 128; off > 0; off >>= 1) {
    if (tid < off) s1[tid] = fmaxf(s1[tid], s1[tid + off]);
    __syncthreads();
  }
  const float scale = 127.f / fmaxf(s1[0], 1e-5f);
  const float inv   = 1.f / scale;
#pragma unroll
  for (int k = 0; k < 4; ++k) {
    float q = rintf(fminf(fmaxf(xn[k] * scale, -127.f), 127.f)) * inv;
    xq[row * DMODEL + tid + k * 256] = f32_to_bf16(q);
  }
}

// ---------------------------------------------------------------------------
// Stage 2b: logits[16(pad)][16384] = xq @ W_l^T via v_wmma_f32_16x16x32_bf16.
// xq (32 KB) is staged once per block into LDS (uint4-granular so clang emits
// ds_store_b128 / ds_load_b128); the global path then carries only the weight
// stream: per WMMA each lane does 2x global_load_b128 (B) + 2x ds_load_b128
// (A). One wave per 16-wide vocab tile, K-loop of 32 WMMAs over D=1024.
// ---------------------------------------------------------------------------
__global__ void __launch_bounds__(256)
gemm_wmma(const unsigned short* __restrict__ Wl,
          const unsigned short* __restrict__ xq,
          float* __restrict__ logits /* [BATCH][VOCAB] */) {
  __shared__ uint4 xs4[16 * DMODEL / 8];               // 32 KB activation tile

  // Cooperative stage: 256 threads x 8 iterations x 16B.
  const uint4* xq4 = (const uint4*)xq;
#pragma unroll
  for (int i = 0; i < 8; ++i)
    xs4[threadIdx.x + i * 256] = xq4[threadIdx.x + i * 256];
  __syncthreads();

  const int lane = threadIdx.x & 31;
  const int wave = threadIdx.x >> 5;
  const int tile = blockIdx.x * 8 + wave;      // 0..1023
  const int v0   = tile * 16;
  const int half = lane >> 4;
  const int r    = lane & 15;

  // A 16x32 bf16: lane<16 row r holds K=k0..k0+7 (V0-3) and k0+16..k0+23 (V4-7);
  //               lane>=16 row r holds K=k0+8..k0+15 and k0+24..k0+31.
  const int abase_lo = (r * DMODEL + half * 8) >> 3;   // uint4 index, +k0/8 later
  const int abase_hi = (r * DMODEL + 16 + half * 8) >> 3;
  // B 32x16 bf16: lane<16 col r holds K=k0..k0+7 and k0+16..k0+23; lane>=16 col r holds K=k0+8..k0+15 and k0+24..k0+31.
  const unsigned short* brow = Wl + ((size_t)(v0 + r)) * DMODEL + half * 8;

  v8f c = {};
#pragma unroll 4
  for (int k0 = 0; k0 < DMODEL; k0 += 32) {
    V16BFCast a, b;
    a.q[0] = xs4[abase_lo + (k0 >> 3)];
    a.q[1] = xs4[abase_hi + (k0 >> 3)];
    b.q[0] = *(const uint4*)(brow + k0);
    b.q[1] = *(const uint4*)(brow + k0 + 16);
    c = __builtin_amdgcn_wmma_f32_16x16x32_bf16(
        /*neg_a=*/false, a.v, /*neg_b=*/false, b.v,
        /*c_mod=*/(short)0, c, /*reuse_a=*/false, /*reuse_b=*/false);
  }

  // C layout: VGPR i, lanes 0-15 -> row M=i, N=lane. Rows 8..15 are padding.
  if (half == 0) {
#pragma unroll
    for (int i = 0; i < 8; ++i)
      logits[(size_t)i * VOCAB + v0 + r] = c[i];
  }
}

// ---------------------------------------------------------------------------
// Stage 2c: deterministic Gumbel-argmax sampling + rolling x update.
// One block per batch row; 256-thread max-reduction over 16384 logits.
// ---------------------------------------------------------------------------
__device__ __forceinline__ unsigned int hash_u32(unsigned int x) {
  x ^= x >> 16; x *= 0x7FEB352Du;
  x ^= x >> 15; x *= 0x846CA68Bu;
  x ^= x >> 16;
  return x;
}

__global__ void __launch_bounds__(256)
sample_shift(const float* __restrict__ logits, float* __restrict__ tok_out,
             const float* __restrict__ xin, float* __restrict__ xout, int t) {
  const int b = blockIdx.x;
  const int tid = threadIdx.x;
  __shared__ float sval[256];
  __shared__ int   sidx[256];
  float best = -3.4e38f; int bi = 0;
  for (int v = tid; v < VOCAB; v += 256) {
    unsigned int h = hash_u32(((unsigned int)((t * BATCH + b) << 16) ^ (unsigned int)v)
                              * 2654435761u + 0x9E3779B9u);
    float u = (float)(h >> 8) * (1.0f / 16777216.0f) + 1e-7f;
    float g = -logf(-logf(u));                           // Gumbel noise
    float val = logits[(size_t)b * VOCAB + v] + g;       // TEMP == 1.0
    if (val > best) { best = val; bi = v; }
  }
  sval[tid] = best; sidx[tid] = bi;
  __syncthreads();
  for (int off = 128; off > 0; off >>= 1) {
    if (tid < off && sval[tid + off] > sval[tid]) {
      sval[tid] = sval[tid + off]; sidx[tid] = sidx[tid + off];
    }
    __syncthreads();
  }
  const int tok = sidx[0];
  for (int d = tid; d < DMODEL - 1; d += 256)
    xout[b * DMODEL + d] = xin[b * DMODEL + d + 1];
  if (tid == 0) {
    xout[b * DMODEL + DMODEL - 1] = (float)tok;
    tok_out[b] = (float)tok;
  }
}

// ---------------------------------------------------------------------------
extern "C" void kernel_launch(void* const* d_in, const int* in_sizes, int n_in,
                              void* d_out, int out_size, void* d_ws, size_t ws_size,
                              hipStream_t stream) {
  (void)in_sizes; (void)n_in; (void)out_size; (void)ws_size;
  const float* x_in   = (const float*)d_in[0];   // [8][1024] f32
  const int*   seeds  = (const int*)d_in[1];     // [4][65536] i32
  const float* scales = (const float*)d_in[2];   // [4] f32
  float* out = (float*)d_out;                    // toks[16][8] ++ logits[16][8][16384]

  // Workspace layout: bf16 weights (134 MB, L2-resident), ping-pong x, xq.
  unsigned short* W = (unsigned short*)d_ws;
  const size_t wbytes = (size_t)NLAYERS * VOCAB * DMODEL * sizeof(unsigned short);
  float* xbuf0 = (float*)((char*)d_ws + wbytes);
  float* xbuf1 = xbuf0 + BATCH * DMODEL;
  unsigned short* xq = (unsigned short*)(xbuf1 + BATCH * DMODEL);

  hipMemcpyAsync(xbuf0, x_in, (size_t)BATCH * DMODEL * sizeof(float),
                 hipMemcpyDeviceToDevice, stream);

  decompress_lfsr<<<NSEEDS / 256, 256, 0, stream>>>(seeds, scales, W);

  float* tok_base   = out;                         // [16][8]
  float* logit_base = out + TSTEPS * BATCH;        // [16][8][16384]

  for (int t = 0; t < TSTEPS; ++t) {
    float* xi = (t & 1) ? xbuf1 : xbuf0;
    float* xo = (t & 1) ? xbuf0 : xbuf1;
    const unsigned short* Wl = W + (size_t)(t & (NLAYERS - 1)) * VOCAB * DMODEL;
    float* lg = logit_base + (size_t)t * BATCH * VOCAB;

    ln_quant<<<16, 256, 0, stream>>>(xi, xq);
    gemm_wmma<<<VOCAB / 16 / 8, 256, 0, stream>>>(Wl, xq, lg);
    sample_shift<<<BATCH, 256, 0, stream>>>(lg, tok_base + t * BATCH, xi, xo, t);
  }
}
